// TreeFilterV2_11982958756205
// MI455X (gfx1250) — matrix operand order repaired and checked by reading, not compiled, not run on hardware
//
#include <hip/hip_runtime.h>
#include <math.h>

// ---- problem constants (match reference) ----
#define BB   4
#define CINC 256
#define CE   64
#define GG   4
#define NN   6400
#define KKD  256
#define EPSF 1e-8f
#define CPG  65      // channels per group in scan payload (64 feat + 1 norm)
#define CTOT 260     // GG * CPG

typedef __bf16 bf16_t;
typedef bf16_t v16bf __attribute__((ext_vector_type(16)));
typedef float  v8f   __attribute__((ext_vector_type(8)));
typedef unsigned int v8u __attribute__((ext_vector_type(8)));

__device__ __forceinline__ unsigned short f32_to_bf16(float f) {
  unsigned int u = __builtin_bit_cast(unsigned int, f);
  u += 0x7FFFu + ((u >> 16) & 1u);     // round-to-nearest-even
  return (unsigned short)(u >> 16);
}

// -----------------------------------------------------------------------------
// GEMM: Y[b] (64 x N) = Wt (64 x 256) * X[b] (256 x N), bf16 WMMA, f32 accum.
// Block = 128 threads (4 waves). Block tile: 64(M) x 64(N). Each wave owns one
// 16-wide n-subtile and all 4 m-tiles. K chunked by 32 through LDS.
// -----------------------------------------------------------------------------
__global__ __launch_bounds__(128)
void tf_gemm64_wmma(const float* __restrict__ Wt,   // (64, 256) row-major
                    const float* __restrict__ X,    // (B, 256, N)
                    float* __restrict__ Y)          // (B, 64, N)
{
  __shared__ unsigned short Wl[64][34];   // [m][kk]  (pad 34 to stagger banks)
  __shared__ unsigned short Xl[64][34];   // [n][kk]  (transposed stage)

  const int t    = threadIdx.x;
  const int lane = t & 31;
  const int wv   = t >> 5;
  const int b    = blockIdx.y;
  const int col0 = blockIdx.x * 64;

  const float* Xb = X + (size_t)b * KKD * NN;

  v8f acc[4];
  #pragma unroll
  for (int i = 0; i < 4; ++i) acc[i] = v8f{};

  const int mrow = lane & 15;   // M (for A) / N (for B) index inside tile
  const int hi   = lane >> 4;   // half-wave selector per ISA layout

  for (int k0 = 0; k0 < KKD; k0 += 32) {
    // Stage W chunk (64 x 32): coalesced along k
    #pragma unroll
    for (int i = 0; i < 16; ++i) {
      int e = i * 128 + t;
      int m = e >> 5, kk = e & 31;
      Wl[m][kk] = f32_to_bf16(Wt[m * KKD + k0 + kk]);
    }
    // Stage X chunk transposed (n x kk): coalesced along n in global
    #pragma unroll
    for (int i = 0; i < 16; ++i) {
      int e  = i * 128 + t;
      int kk = e >> 6, n = e & 63;
      Xl[n][kk] = f32_to_bf16(Xb[(size_t)(k0 + kk) * NN + col0 + n]);
    }
    __syncthreads();

    // B fragment (32x16): lane = N; VGPR0..7 hold K=hi*16 .. hi*16+15 packed.
    v8u bu;
    {
      const unsigned int* q =
          (const unsigned int*)(&Xl[wv * 16 + mrow][0] + hi * 16);
      #pragma unroll
      for (int j = 0; j < 8; ++j) bu[j] = q[j];
    }
    v16bf bfrag = __builtin_bit_cast(v16bf, bu);

    #pragma unroll
    for (int mt = 0; mt < 4; ++mt) {
      // A fragment (16x32): lane = M; VGPR0..3: K=hi*16..+7, VGPR4..7: K=hi*16+8..+15
      v8u au;
      const unsigned short* row = &Wl[mt * 16 + mrow][0];
      const unsigned int* p0 = (const unsigned int*)(row + hi * 16);
      const unsigned int* p1 = (const unsigned int*)(row + hi * 16 + 8);
      #pragma unroll
      for (int j = 0; j < 4; ++j) { au[j] = p0[j]; au[4 + j] = p1[j]; }
      v16bf afrag = __builtin_bit_cast(v16bf, au);

      acc[mt] = __builtin_amdgcn_wmma_f32_16x16x32_bf16(
          false, afrag, false, bfrag, (short)0, acc[mt], false, false);
    }
    __syncthreads();
  }

  // C/D layout: VGPR r -> M = hi*8 + r, N = lane&15
  float* Yb = Y + (size_t)b * CE * NN;
  const int n = col0 + wv * 16 + mrow;
  #pragma unroll
  for (int mt = 0; mt < 4; ++mt) {
    #pragma unroll
    for (int r = 0; r < 8; ++r) {
      int m = mt * 16 + hi * 8 + r;
      Yb[(size_t)m * NN + n] = acc[mt][r];
    }
  }
}

// -----------------------------------------------------------------------------
// conf = sigmoid(W_conf @ f): one thread per (b, n), 4 dot products, coalesced.
// -----------------------------------------------------------------------------
__global__ void tf_conf(const float* __restrict__ F,
                        const float* __restrict__ Wc,
                        float* __restrict__ conf)
{
  int e = blockIdx.x * blockDim.x + threadIdx.x;
  if (e >= BB * NN) return;
  int b = e / NN, n = e % NN;
  const float* Fb = F + (size_t)b * CINC * NN;
  float a0 = 0.f, a1 = 0.f, a2 = 0.f, a3 = 0.f;
  for (int c = 0; c < CINC; ++c) {
    float fv = Fb[(size_t)c * NN + n];
    a0 += Wc[0 * CINC + c] * fv;
    a1 += Wc[1 * CINC + c] * fv;
    a2 += Wc[2 * CINC + c] * fv;
    a3 += Wc[3 * CINC + c] * fv;
  }
  float* cb = conf + (size_t)b * GG * NN;
  cb[0 * NN + n] = 1.f / (1.f + expf(-a0));
  cb[1 * NN + n] = 1.f / (1.f + expf(-a1));
  cb[2 * NN + n] = 1.f / (1.f + expf(-a2));
  cb[3 * NN + n] = 1.f / (1.f + expf(-a3));
}

// -----------------------------------------------------------------------------
// Edge weights: w[b][j][g] = exp(-(d_e + d_g + beta[g]^2)), gathered via order.
// -----------------------------------------------------------------------------
__global__ void tf_edge_w(const float* __restrict__ E,
                          const float* __restrict__ Ge,
                          const float* __restrict__ beta,
                          const int* __restrict__ order,
                          const int* __restrict__ parent,
                          float* __restrict__ w)
{
  int e = blockIdx.x * blockDim.x + threadIdx.x;
  if (e >= BB * NN) return;
  int b = e / NN, j = e % NN;
  int node  = order[b * NN + j];
  int pj    = parent[b * NN + j];
  int pnode = order[b * NN + pj];
  const float* Eb = E  + (size_t)b * CE * NN;
  const float* Gb = Ge + (size_t)b * CE * NN;
  float* wb = w + ((size_t)b * NN + j) * GG;
  #pragma unroll
  for (int g = 0; g < GG; ++g) {
    float d = 0.f;
    #pragma unroll
    for (int cc = 0; cc < 16; ++cc) {
      int c = g * 16 + cc;
      float de = Eb[(size_t)c * NN + node] - Eb[(size_t)c * NN + pnode];
      d += de * de;
      float dg = Gb[(size_t)c * NN + node] - Gb[(size_t)c * NN + pnode];
      d += dg * dg;
    }
    wb[g] = expf(-(d + beta[g] * beta[g]));
  }
}

// -----------------------------------------------------------------------------
// Build scan payload buf[b][j][g*65+cc] = (cc<64 ? f[g*64+cc]*conf : conf) at
// node order[j]. Writes fully coalesced.
// -----------------------------------------------------------------------------
__global__ void tf_build_xs(const float* __restrict__ F,
                            const float* __restrict__ conf,
                            const int* __restrict__ order,
                            float* __restrict__ buf)
{
  size_t e = (size_t)blockIdx.x * blockDim.x + threadIdx.x;
  if (e >= (size_t)BB * NN * CTOT) return;
  int b  = (int)(e / ((size_t)NN * CTOT));
  int r  = (int)(e % ((size_t)NN * CTOT));
  int j  = r / CTOT;
  int ch = r % CTOT;
  int g  = ch / CPG;
  int cc = ch % CPG;
  int node = order[b * NN + j];
  float cf = conf[((size_t)b * GG + g) * NN + node];
  float v  = (cc < 64) ? F[((size_t)b * CINC + g * 64 + cc) * NN + node] * cf : cf;
  buf[e] = v;
}

// -----------------------------------------------------------------------------
// Tree filter scans. One thread per (batch, channel column): columns are fully
// independent, so each thread runs the whole sequential up+down sweep with no
// barriers. Lanes share k -> every step is one coalesced load + coalesced RMW,
// and the 6.6MB/batch working set is L2-resident. Down sweep runs in place.
// -----------------------------------------------------------------------------
__global__ __launch_bounds__(288)
void tf_tree_scan(const float* __restrict__ w,
                  const int* __restrict__ parent,
                  float* __restrict__ buf)
{
  const int b  = blockIdx.x;
  const int ch = threadIdx.x;
  if (ch >= CTOT) return;
  const int g = ch / CPG;
  const int*  par = parent + b * NN;
  const float* wb = w + (size_t)b * NN * GG;
  float* Ab = buf + (size_t)b * NN * CTOT;

  // Up sweep: agg[parent[k]] += w[k] * agg[k], k = N-1 .. 1
  for (int k = NN - 1; k >= 1; --k) {
    if ((k & 63) == 0 && k > 256)
      __builtin_prefetch(wb + (size_t)(k - 256) * GG, 0, 1);  // global_prefetch_b8
    int   p  = par[k];
    float wk = wb[(size_t)k * GG + g];
    float a  = Ab[(size_t)k * CTOT + ch];
    Ab[(size_t)p * CTOT + ch] += wk * a;
  }
  // Down sweep (in place): out[k] = agg[k] + w[k]*(out[p] - w[k]*agg[k])
  for (int k = 1; k < NN; ++k) {
    int   p  = par[k];
    float wk = wb[(size_t)k * GG + g];
    float a  = Ab[(size_t)k * CTOT + ch];
    float o  = Ab[(size_t)p * CTOT + ch];
    Ab[(size_t)k * CTOT + ch] = a + wk * (o - wk * a);
  }
}

// -----------------------------------------------------------------------------
// Scatter back through the permutation and apply residual:
// out[b][c][order[j]] = gamma * filt/(eps+norm) + f[b][c][order[j]]
// -----------------------------------------------------------------------------
__global__ void tf_finalize(const float* __restrict__ buf,
                            const float* __restrict__ F,
                            const int* __restrict__ order,
                            const float* __restrict__ gamma,
                            float* __restrict__ out)
{
  size_t e = (size_t)blockIdx.x * blockDim.x + threadIdx.x;
  if (e >= (size_t)BB * CINC * NN) return;
  int b = (int)(e / ((size_t)CINC * NN));
  int r = (int)(e % ((size_t)CINC * NN));
  int c = r / NN;
  int j = r % NN;
  int node = order[b * NN + j];
  int g = c >> 6, cc = c & 63;
  const float* Ab = buf + ((size_t)b * NN + j) * CTOT;
  float filt = Ab[g * CPG + cc];
  float norm = Ab[g * CPG + 64];
  float fv   = F[((size_t)b * CINC + c) * NN + node];
  out[((size_t)b * CINC + c) * NN + node] = gamma[0] * (filt / (EPSF + norm)) + fv;
}

// -----------------------------------------------------------------------------
extern "C" void kernel_launch(void* const* d_in, const int* in_sizes, int n_in,
                              void* d_out, int out_size, void* d_ws, size_t ws_size,
                              hipStream_t stream) {
  (void)in_sizes; (void)n_in; (void)out_size;

  const float* feature = (const float*)d_in[0];
  const float* guide   = (const float*)d_in[1];
  const float* W_embed = (const float*)d_in[2];
  const float* W_conf  = (const float*)d_in[3];
  const float* W_guide = (const float*)d_in[4];
  const float* beta    = (const float*)d_in[5];
  const float* gamma   = (const float*)d_in[6];
  const int*   order   = (const int*)d_in[7];
  const int*   parent  = (const int*)d_in[8];
  float* out = (float*)d_out;

  char* ws = (char*)d_ws;
  size_t off = 0;
  float* embed = (float*)(ws + off); off += (size_t)BB * CE * NN * sizeof(float);
  float* gemb  = (float*)(ws + off); off += (size_t)BB * CE * NN * sizeof(float);
  float* conf  = (float*)(ws + off); off += (size_t)BB * GG * NN * sizeof(float);
  float* w     = (float*)(ws + off); off += (size_t)BB * NN * GG * sizeof(float);
  float* buf   = (float*)(ws + off); off += (size_t)BB * NN * CTOT * sizeof(float);
  if (off > ws_size) return;   // workspace too small; nothing safe to do

  dim3 gemm_grid(NN / 64, BB);
  tf_gemm64_wmma<<<gemm_grid, 128, 0, stream>>>(W_embed, feature, embed);
  tf_gemm64_wmma<<<gemm_grid, 128, 0, stream>>>(W_guide, guide,   gemb);

  int bn = (BB * NN + 255) / 256;
  tf_conf  <<<bn, 256, 0, stream>>>(feature, W_conf, conf);
  tf_edge_w<<<bn, 256, 0, stream>>>(embed, gemb, beta, order, parent, w);

  size_t xs_total = (size_t)BB * NN * CTOT;
  tf_build_xs<<<(unsigned)((xs_total + 255) / 256), 256, 0, stream>>>(feature, conf, order, buf);

  tf_tree_scan<<<BB, 288, 0, stream>>>(w, parent, buf);

  size_t out_total = (size_t)BB * CINC * NN;
  tf_finalize<<<(unsigned)((out_total + 255) / 256), 256, 0, stream>>>(buf, feature, order, gamma, out);
}
